// MultiHeadAttention_24584392802641
// MI455X (gfx1250) — matrix operand order repaired and checked by reading, not compiled, not run on hardware
//
#include <hip/hip_runtime.h>
#include <hip/hip_bf16.h>

typedef __attribute__((ext_vector_type(8)))  float   v8f;
typedef __attribute__((ext_vector_type(8)))  __bf16  v8bf;
typedef __attribute__((ext_vector_type(16))) __bf16  v16bf;

constexpr int   Bc   = 4;
constexpr int   Sc   = 2048;
constexpr int   Dc   = 1024;
constexpr int   Hc   = 16;
constexpr int   HSc  = 64;
constexpr int   Mtot = Bc * Sc;        // 8192
constexpr float SCALE = 0.125f;        // 1/sqrt(64)

__device__ __forceinline__ __bf16 f2bf(float f) {
  __hip_bfloat16 h = __float2bfloat16(f);
  __bf16 r;
  __builtin_memcpy(&r, &h, sizeof(r));
  return r;
}

__device__ __forceinline__ v16bf join16(v8bf lo, v8bf hi) {
  return __builtin_shufflevector(lo, hi, 0, 1, 2, 3, 4, 5, 6, 7,
                                 8, 9, 10, 11, 12, 13, 14, 15);
}

// A-fragment (16x32 bf16, lane = row M):
//  a[i] = A[row][k0 + half*16 + i]
__device__ __forceinline__ v16bf load_a_frag(const __bf16* rowp, int k0, int half) {
  const __bf16* p0 = rowp + k0 + half * 16;
  v8bf lo = *(const v8bf*)p0;
  v8bf hi = *(const v8bf*)(p0 + 8);
  return join16(lo, hi);
}

// CDNA5 async copy: 16B global -> LDS, tracked on ASYNCcnt (not LOADcnt).
// Flat address of a __shared__ object carries the LDS byte offset in bits[31:0].
__device__ __forceinline__ void async_copy_b128(void* lds_dst, const void* g_src) {
  uint32_t l = (uint32_t)(uintptr_t)lds_dst;
  asm volatile("global_load_async_to_lds_b128 %0, %1, off"
               :: "v"(l), "v"(g_src) : "memory");
}
__device__ __forceinline__ void wait_async() {
  asm volatile("s_wait_asynccnt 0x0" ::: "memory");
}

// ---------------------------------------------------------------------------
// Prep: fp32 -> bf16 cast, and transposing weight casts
// ---------------------------------------------------------------------------
__global__ void cast_kernel(const float* __restrict__ in, __bf16* __restrict__ out, int n) {
  int i = blockIdx.x * blockDim.x + threadIdx.x;
  if (i < n) out[i] = f2bf(in[i]);
}

// in[h][r][c] -> out[h][c][r]
__global__ void transpose_cast_kernel(const float* __restrict__ in, __bf16* __restrict__ out,
                                      int Hn, int R, int C) {
  int i = blockIdx.x * blockDim.x + threadIdx.x;
  int total = Hn * R * C;
  if (i >= total) return;
  int c = i % C;
  int r = (i / C) % R;
  int h = i / (R * C);
  out[((size_t)h * C + c) * R + r] = f2bf(in[i]);
}

// ---------------------------------------------------------------------------
// WMMA GEMM: C[m][n] = A[m][:] . Bt[n][:] + bias[n]   (bf16 x bf16 -> f32)
// Block: 256 threads = 8 waves; tile 256(M) x 64(N); wave = 32 x 64 strip.
// B staged through LDS with async-to-LDS, double-buffered in 128-K slabs.
// ---------------------------------------------------------------------------
enum GemmMode { MODE_QK = 0, MODE_VT = 1, MODE_OUT = 2 };

template <int MODE>
__global__ __launch_bounds__(256) void gemm_proj(
    const __bf16* __restrict__ A,    // [Mtot, K] row-major
    const __bf16* __restrict__ Bt,   // [N, K] row-major (B transposed)
    const float* __restrict__ bias,  // [N]
    void* __restrict__ Cout,
    float scale) {
  constexpr int K  = Dc;   // 1024
  constexpr int KB = 128;  // K slab staged per buffer
  __shared__ alignas(16) __bf16 ldsB[2][64][136];  // +8 pad: 16 distinct banks

  const int tid  = threadIdx.x;
  const int lane = tid & 31;
  const int wave = tid >> 5;
  const int row  = lane & 15;
  const int half = lane >> 4;
  const int m0 = blockIdx.x * 256 + wave * 32;
  const int n0 = blockIdx.y * 64;

  const __bf16* arow0 = A + (size_t)(m0 + row) * K;
  const __bf16* arow1 = A + (size_t)(m0 + 16 + row) * K;

  auto stage = [&](int buf, int kb) {
#pragma unroll
    for (int j = 0; j < 4; ++j) {
      int c = tid + 256 * j;          // 1024 chunks of 16B = 64 x 256B rows
      int r = c >> 4, cc = c & 15;
      async_copy_b128(&ldsB[buf][r][cc * 8],
                      Bt + (size_t)(n0 + r) * K + kb * KB + cc * 8);
    }
  };

  v8f acc[2][4] = {};

  stage(0, 0);
  wait_async();
  __syncthreads();

  for (int kb = 0; kb < K / KB; ++kb) {
    const int cur = kb & 1;
    if (kb + 1 < K / KB) stage(cur ^ 1, kb + 1);

#pragma unroll
    for (int kc = 0; kc < 4; ++kc) {
      const int k0 = kb * KB + kc * 32;
      // Issue ALL fragment loads for this k-chunk first, then the WMMA chain,
      // so DS latency is covered by counter-threshold waits, not dscnt==0.
      v16bf bfrag[4];
#pragma unroll
      for (int nt = 0; nt < 4; ++nt)
        bfrag[nt] = *(const v16bf*)&ldsB[cur][nt * 16 + row][kc * 32 + half * 16];
      v16bf af0 = load_a_frag(arow0, k0, half);
      v16bf af1 = load_a_frag(arow1, k0, half);
#pragma unroll
      for (int nt = 0; nt < 4; ++nt) {
        acc[0][nt] = __builtin_amdgcn_wmma_f32_16x16x32_bf16(
            false, af0, false, bfrag[nt], (short)0, acc[0][nt], false, false);
        acc[1][nt] = __builtin_amdgcn_wmma_f32_16x16x32_bf16(
            false, af1, false, bfrag[nt], (short)0, acc[1][nt], false, false);
      }
    }
    wait_async();
    __syncthreads();
  }

  // Epilogue. C/D layout: col N = lane&15, row M = r + 8*half.
#pragma unroll
  for (int mt = 0; mt < 2; ++mt) {
#pragma unroll
    for (int nt = 0; nt < 4; ++nt) {
      const int n = n0 + nt * 16 + row;
      const float bn = bias[n];
#pragma unroll
      for (int r = 0; r < 8; ++r) {
        const int m = m0 + mt * 16 + r + half * 8;
        float v = (acc[mt][nt][r] + bn) * scale;
        if (MODE == MODE_OUT) {
          ((float*)Cout)[(size_t)m * Dc + n] = v;
        } else {
          const int b = m / Sc, s = m % Sc, h = n >> 6, e = n & 63;
          __bf16* out = (__bf16*)Cout;
          if (MODE == MODE_QK)
            out[(((size_t)b * Hc + h) * Sc + s) * HSc + e] = f2bf(v);   // [B,H,S,HS]
          else
            out[(((size_t)b * Hc + h) * HSc + e) * Sc + s] = f2bf(v);   // [B,H,HS,S]
        }
      }
    }
  }
}

// ---------------------------------------------------------------------------
// Flash attention forward. Grid: (S/128, H, B); 8 waves; wave = 16 query rows.
// K/V tiles (32 keys) staged once per block via async-to-LDS, double-buffered.
// Q pre-scaled by 1/sqrt(HS). Online softmax with causal mask.
// ---------------------------------------------------------------------------
__global__ __launch_bounds__(256) void attn_fwd(
    const __bf16* __restrict__ Q,    // [B,H,S,HS] (scaled)
    const __bf16* __restrict__ Kt,   // [B,H,S,HS]
    const __bf16* __restrict__ Vt,   // [B,H,HS,S]
    __bf16* __restrict__ Out) {      // [B,S,H*HS]
  __shared__ alignas(16) __bf16 ldsK[2][32][72];   // [key][hs], padded
  __shared__ alignas(16) __bf16 ldsV[2][64][40];   // [hs][key], padded
  __shared__ alignas(16) __bf16 ldsP[8][16][40];   // per-wave P tile

  const int tid  = threadIdx.x;
  const int lane = tid & 31;
  const int wave = tid >> 5;
  const int row  = lane & 15;
  const int half = lane >> 4;
  const int b = blockIdx.z, h = blockIdx.y;
  const int qr = blockIdx.x * 128 + wave * 16;

  const size_t bh = (size_t)b * Hc + h;
  const __bf16* Qbh = Q + bh * Sc * HSc;
  const __bf16* Kbh = Kt + bh * Sc * HSc;
  const __bf16* Vbh = Vt + bh * HSc * Sc;

  auto stage = [&](int buf, int kb) {
    const int kbase = kb * 32;
    {  // K tile: 32 rows x 64 elems = 256 x 16B chunks
      int r = tid >> 3, c = tid & 7;
      async_copy_b128(&ldsK[buf][r][c * 8],
                      Kbh + (size_t)(kbase + r) * HSc + c * 8);
    }
    {  // V tile: 64 rows x 32 elems = 256 x 16B chunks
      int r = tid >> 2, c = tid & 3;
      async_copy_b128(&ldsV[buf][r][c * 8],
                      Vbh + (size_t)r * Sc + kbase + c * 8);
    }
  };

  // Q A-fragments for the two K-chunks of HS=64
  v16bf qa[2];
#pragma unroll
  for (int kc = 0; kc < 2; ++kc)
    qa[kc] = load_a_frag(Qbh + (size_t)(qr + row) * HSc, kc * 32, half);

  v8f o[4] = {};
  float mrow[8], lrow[8];
#pragma unroll
  for (int r = 0; r < 8; ++r) { mrow[r] = -3.0e38f; lrow[r] = 0.0f; }

  const int kbMaxWave  = (qr + 16 + 31) >> 5;   // this wave's causal edge
  const int kbMaxBlock = (blockIdx.x + 1) * 4;  // block-uniform loop bound

  stage(0, 0);
  wait_async();
  __syncthreads();

  for (int kb = 0; kb < kbMaxBlock; ++kb) {
    const int cur = kb & 1;
    if (kb + 1 < kbMaxBlock) stage(cur ^ 1, kb + 1);

    if (kb < kbMaxWave) {  // wave-uniform guard; all waves still hit barriers
      const int kbase = kb * 32;

      // Load ALL K fragments for this key block, then issue the WMMA chain.
      v16bf kf[2][2];
#pragma unroll
      for (int nt = 0; nt < 2; ++nt)
#pragma unroll
        for (int kc = 0; kc < 2; ++kc)
          kf[nt][kc] = *(const v16bf*)&ldsK[cur][nt * 16 + row][kc * 32 + half * 16];

      v8f sc[2];
#pragma unroll
      for (int nt = 0; nt < 2; ++nt) {
        v8f t = {};
#pragma unroll
        for (int kc = 0; kc < 2; ++kc)
          t = __builtin_amdgcn_wmma_f32_16x16x32_bf16(
              false, qa[kc], false, kf[nt][kc], (short)0, t, false, false);
        sc[nt] = t;
      }

      // causal mask + online softmax; C rows = r + 8*half, C col = lane&15
      const int col0 = kbase + row;
#pragma unroll
      for (int r = 0; r < 8; ++r) {
        const int crow = qr + half * 8 + r;
        sc[0][r] = (col0 > crow) ? -1.0e9f : sc[0][r];
        sc[1][r] = (col0 + 16 > crow) ? -1.0e9f : sc[1][r];

        float mx = fmaxf(sc[0][r], sc[1][r]);
#pragma unroll
        for (int off = 1; off < 16; off <<= 1)
          mx = fmaxf(mx, __shfl_xor(mx, off, 16));
        const float mnew = fmaxf(mrow[r], mx);
        const float corr = __expf(mrow[r] - mnew);
        const float p0 = __expf(sc[0][r] - mnew);
        const float p1 = __expf(sc[1][r] - mnew);
        float rs = p0 + p1;
#pragma unroll
        for (int off = 1; off < 16; off <<= 1)
          rs += __shfl_xor(rs, off, 16);
        lrow[r] = lrow[r] * corr + rs;
        mrow[r] = mnew;
#pragma unroll
        for (int nt = 0; nt < 4; ++nt) o[nt][r] *= corr;

        const int prow = half * 8 + r;
        ldsP[wave][prow][row]      = f2bf(p0);
        ldsP[wave][prow][16 + row] = f2bf(p1);
      }

      // Transposed read of P as an A-fragment, plus all V fragments, then WMMAs.
      const __bf16* pr = &ldsP[wave][row][0];
      v8bf plo = *(const v8bf*)(pr + half * 16);
      v8bf phi = *(const v8bf*)(pr + half * 16 + 8);
      v16bf vf[4];
#pragma unroll
      for (int nt = 0; nt < 4; ++nt)
        vf[nt] = *(const v16bf*)&ldsV[cur][nt * 16 + row][half * 16];
      v16bf pf = join16(plo, phi);

#pragma unroll
      for (int nt = 0; nt < 4; ++nt)
        o[nt] = __builtin_amdgcn_wmma_f32_16x16x32_bf16(
            false, pf, false, vf[nt], (short)0, o[nt], false, false);
    }

    wait_async();
    __syncthreads();
  }

  // normalize and write concatenated heads [B,S,H*HS] as bf16
#pragma unroll
  for (int nt = 0; nt < 4; ++nt) {
    const int n = h * HSc + nt * 16 + row;
#pragma unroll
    for (int r = 0; r < 8; ++r) {
      const int s = qr + half * 8 + r;
      const float v = o[nt][r] / lrow[r];
      Out[((size_t)b * Sc + s) * (size_t)Dc + n] = f2bf(v);
    }
  }
}

// ---------------------------------------------------------------------------
// Launch
// ---------------------------------------------------------------------------
extern "C" void kernel_launch(void* const* d_in, const int* in_sizes, int n_in,
                              void* d_out, int out_size, void* d_ws, size_t ws_size,
                              hipStream_t stream) {
  const float* x  = (const float*)d_in[0];
  const float* Wq = (const float*)d_in[1];
  const float* bq = (const float*)d_in[2];
  const float* Wk = (const float*)d_in[3];
  const float* bk = (const float*)d_in[4];
  const float* Wv = (const float*)d_in[5];
  const float* bv = (const float*)d_in[6];
  const float* Wo = (const float*)d_in[7];
  const float* bo = (const float*)d_in[8];

  char* ws = (char*)d_ws;
  size_t off = 0;
  auto carve = [&](size_t elems) -> __bf16* {
    __bf16* p = (__bf16*)(ws + off);
    off += ((elems * sizeof(__bf16) + 255) / 256) * 256;
    return p;
  };
  __bf16* xb    = carve((size_t)Mtot * Dc);            // x in bf16
  __bf16* Wqt   = carve((size_t)Hc * HSc * Dc);        // [n][k]
  __bf16* Wkt   = carve((size_t)Hc * HSc * Dc);
  __bf16* Wvt   = carve((size_t)Hc * HSc * Dc);
  __bf16* WoT   = carve((size_t)Dc * Hc * HSc);        // [n][k]
  __bf16* qb    = carve((size_t)Bc * Hc * Sc * HSc);   // [B,H,S,HS] scaled
  __bf16* kbuf  = carve((size_t)Bc * Hc * Sc * HSc);   // [B,H,S,HS]
  __bf16* vtb   = carve((size_t)Bc * Hc * HSc * Sc);   // [B,H,HS,S]
  __bf16* attnb = carve((size_t)Mtot * Hc * HSc);      // [B,S,H*HS]

  int n = Mtot * Dc;
  cast_kernel<<<(n + 255) / 256, 256, 0, stream>>>(x, xb, n);
  n = Hc * Dc * HSc;
  transpose_cast_kernel<<<(n + 255) / 256, 256, 0, stream>>>(Wq, Wqt, Hc, Dc, HSc);
  transpose_cast_kernel<<<(n + 255) / 256, 256, 0, stream>>>(Wk, Wkt, Hc, Dc, HSc);
  transpose_cast_kernel<<<(n + 255) / 256, 256, 0, stream>>>(Wv, Wvt, Hc, Dc, HSc);
  n = Dc * Hc * HSc;
  transpose_cast_kernel<<<(n + 255) / 256, 256, 0, stream>>>(Wo, WoT, 1, Hc * HSc, Dc);

  dim3 gg(Mtot / 256, (Hc * HSc) / 64), bb(256, 1, 1);
  gemm_proj<MODE_QK><<<gg, bb, 0, stream>>>(xb, Wqt, bq, (void*)qb, SCALE);
  gemm_proj<MODE_QK><<<gg, bb, 0, stream>>>(xb, Wkt, bk, (void*)kbuf, 1.0f);
  gemm_proj<MODE_VT><<<gg, bb, 0, stream>>>(xb, Wvt, bv, (void*)vtb, 1.0f);

  attn_fwd<<<dim3(Sc / 128, Hc, Bc), 256, 0, stream>>>(qb, kbuf, vtb, attnb);

  gemm_proj<MODE_OUT><<<gg, bb, 0, stream>>>(attnb, WoT, bo, d_out, 1.0f);

  (void)in_sizes; (void)n_in; (void)out_size; (void)ws_size;
}
